// NTMCell_86517821211573
// MI455X (gfx1250) — compile-verified, matrix operand order checked by reading, not run on hardware
//
#include <hip/hip_runtime.h>
#include <hip/hip_bf16.h>
#include <math.h>

// NTM cell forward for MI455X (gfx1250, wave32).
// Memory-bound: C is 256MB; min traffic 2x read + 1x write = 768MB ~= 33us @ 23.3TB/s.
// Flash-style online softmax fuses dots/norms/softmax/weighted-read into ONE pass over C.
// Dense h@W.T GEMMs run on v_wmma_f32_16x16x32_f16 with pre-converted f16 operands
// (b128 fragment loads, no per-tile cvt).

#define N_B   128
#define DH    512
#define M_SL  4096
#define D_W   128
#define EPSF  1e-8f

typedef __attribute__((ext_vector_type(16))) _Float16 v16h;
typedef __attribute__((ext_vector_type(8)))  _Float16 v8h;
typedef __attribute__((ext_vector_type(8)))  float    v8f;
typedef __attribute__((ext_vector_type(4)))  float    v4f;

// ---------------------------------------------------------------------------
// f32 -> f16 conversion (one-time, for weights / h_prev)
// ---------------------------------------------------------------------------
__global__ void cvt_f16_kernel(const float* __restrict__ in, _Float16* __restrict__ out,
                               int n) {
  const int i = blockIdx.x * blockDim.x + threadIdx.x;
  if (i < n) out[i] = (_Float16)in[i];
}

// ---------------------------------------------------------------------------
// Y[Nrows, Ncols] = X[Nrows, K] @ W[Ncols, K]^T + bias[Ncols]   (X, W in f16)
// One wave per 16x16 output tile, f32 accumulate.
// A-frag (16x32 f16, ISA 7.12.2): lane<16 -> M=lane, K={kb..kb+7, kb+16..kb+23}, kb=0
//                                  lane>=16 -> same pattern with kb=8.
// B-frag mirrors A with N=lane&15 (B = W^T tile, so read rows of W).
// Fragment = two contiguous v8h groups -> two global_load_b128 per operand.
// C/D frag: n = lane&15, m = vgpr + 8*(lane>>4).
// ---------------------------------------------------------------------------
__global__ void gemm_xwt_wmma(const _Float16* __restrict__ X, const _Float16* __restrict__ W,
                              const float* __restrict__ bias, float* __restrict__ Y,
                              int Nrows, int Ncols, int K) {
  const int lane = threadIdx.x & 31;
  const int wib  = threadIdx.x >> 5;
  const int wpb  = blockDim.x >> 5;
  const int tilesX = Ncols >> 4;
  const int tilesY = Nrows >> 4;
  const int tile = blockIdx.x * wpb + wib;
  if (tile >= tilesX * tilesY) return;            // wave-uniform: EXEC stays all-1s
  const int tr = tile / tilesX;
  const int tc = tile % tilesX;

  const int r16 = lane & 15;
  const int kb  = (lane >> 4) * 8;                // 0 or 8

  const _Float16* __restrict__ xr = X + (size_t)(tr * 16 + r16) * K + kb;
  const _Float16* __restrict__ wr = W + (size_t)(tc * 16 + r16) * K + kb;

  v8f acc = {};
  for (int k0 = 0; k0 < K; k0 += 64) {            // K % 64 == 0 for all our GEMMs
#pragma unroll
    for (int u = 0; u < 2; ++u) {
      const int kk = k0 + u * 32;
      const v8h alo = *(const v8h*)(xr + kk);
      const v8h ahi = *(const v8h*)(xr + kk + 16);
      const v8h blo = *(const v8h*)(wr + kk);
      const v8h bhi = *(const v8h*)(wr + kk + 16);
      const v16h a = __builtin_shufflevector(alo, ahi, 0, 1, 2, 3, 4, 5, 6, 7,
                                             8, 9, 10, 11, 12, 13, 14, 15);
      const v16h b = __builtin_shufflevector(blo, bhi, 0, 1, 2, 3, 4, 5, 6, 7,
                                             8, 9, 10, 11, 12, 13, 14, 15);
      acc = __builtin_amdgcn_wmma_f32_16x16x32_f16(false, a, false, b,
                                                   (short)0, acc, false, false);
    }
  }

  const int n     = tc * 16 + r16;
  const int mbase = tr * 16 + 8 * (lane >> 4);
  const float bv  = bias[n];
#pragma unroll
  for (int vg = 0; vg < 8; ++vg)
    Y[(size_t)(mbase + vg) * Ncols + n] = acc[vg] + bv;
}

// ---------------------------------------------------------------------------
// beta[n] = 1 + softplus(h[n] . Wb + bb)
// ---------------------------------------------------------------------------
__global__ void beta_kernel(const float* __restrict__ h, const float* __restrict__ Wb,
                            const float* __restrict__ bb, float* __restrict__ beta) {
  __shared__ float red[128];
  const int n = blockIdx.x, t = threadIdx.x;
  float p = 0.f;
  for (int j = t; j < DH; j += 128) p += h[n * DH + j] * Wb[j];
  red[t] = p; __syncthreads();
  for (int s = 64; s; s >>= 1) { if (t < s) red[t] += red[t + s]; __syncthreads(); }
  if (t == 0) {
    const float x  = red[0] + bb[0];
    const float sp = (x > 20.f) ? x : log1pf(__expf(x));
    beta[n] = 1.f + sp;
  }
}

// ---------------------------------------------------------------------------
// Flash pass: one block per batch n (256 threads = 8 waves).
// Streams C[n] once: per row m computes dot(C_m,k), ||C_m||, logit s = beta*cos,
// maintains online softmax (max, denom) and weighted accumulation of C rows.
// Outputs: s[n,m] logits, rowmax[n], rowsum[n], r[n,:] (f32 to d_out, f16 to ws).
// ---------------------------------------------------------------------------
__global__ void flash_read_kernel(const float* __restrict__ C, const float* __restrict__ k,
                                  const float* __restrict__ beta,
                                  float* __restrict__ s_out, float* __restrict__ rowmax,
                                  float* __restrict__ rowsum,
                                  float* __restrict__ r_out, _Float16* __restrict__ r16) {
  __shared__ float wm[8], wl[8], wacc[8][D_W];
  const int n    = blockIdx.x;
  const int lane = threadIdx.x & 31;
  const int wave = threadIdx.x >> 5;
  const float* __restrict__ Cn = C + (size_t)n * M_SL * D_W;

  // k[n] and ||k||: 32 lanes x v4f = 128 = D
  const v4f kv = ((const v4f*)(k + n * D_W))[lane];
  float kn2 = kv.x * kv.x + kv.y * kv.y + kv.z * kv.z + kv.w * kv.w;
#pragma unroll
  for (int off = 16; off; off >>= 1) kn2 += __shfl_xor(kn2, off, 32);
  const float k_norm = sqrtf(kn2);
  const float bta    = beta[n];

  float m_run = -INFINITY, l_run = 0.f;
  v4f acc = {0.f, 0.f, 0.f, 0.f};

  for (int m = wave; m < M_SL; m += 8) {
    const v4f cv = ((const v4f*)(Cn + (size_t)m * D_W))[lane];
    float dot = cv.x * kv.x + cv.y * kv.y + cv.z * kv.z + cv.w * kv.w;
    float nrm = cv.x * cv.x + cv.y * cv.y + cv.z * cv.z + cv.w * cv.w;
#pragma unroll
    for (int off = 16; off; off >>= 1) {
      dot += __shfl_xor(dot, off, 32);
      nrm += __shfl_xor(nrm, off, 32);
    }
    const float s = bta * dot / fmaxf(sqrtf(nrm) * k_norm, EPSF);
    if (lane == 0) s_out[(size_t)n * M_SL + m] = s;

    const float m_new = fmaxf(m_run, s);
    const float scale = __expf(m_run - m_new);
    const float p     = __expf(s - m_new);
    l_run = l_run * scale + p;
    acc.x = acc.x * scale + p * cv.x;
    acc.y = acc.y * scale + p * cv.y;
    acc.z = acc.z * scale + p * cv.z;
    acc.w = acc.w * scale + p * cv.w;
    m_run = m_new;
  }

  if (lane == 0) wm[wave] = m_run;
  __syncthreads();
  float gmax = wm[0];
#pragma unroll
  for (int w = 1; w < 8; ++w) gmax = fmaxf(gmax, wm[w]);
  const float f = __expf(m_run - gmax);
  if (lane == 0) wl[wave] = l_run * f;
  wacc[wave][lane * 4 + 0] = acc.x * f;
  wacc[wave][lane * 4 + 1] = acc.y * f;
  wacc[wave][lane * 4 + 2] = acc.z * f;
  wacc[wave][lane * 4 + 3] = acc.w * f;
  __syncthreads();

  if (threadIdx.x < D_W) {
    float L = 0.f, sum = 0.f;
#pragma unroll
    for (int w = 0; w < 8; ++w) { L += wl[w]; sum += wacc[w][threadIdx.x]; }
    const float rv = sum / L;
    r_out[n * D_W + threadIdx.x] = rv;
    r16  [n * D_W + threadIdx.x] = (_Float16)rv;
    if (threadIdx.x == 0) { rowmax[n] = gmax; rowsum[n] = L; }
  }
}

// ---------------------------------------------------------------------------
// GRU elementwise: h' = (1-z)*n + z*h   over [N, DH]; also emits f16 copy.
// ---------------------------------------------------------------------------
__global__ void gru_kernel(const float* __restrict__ gi, const float* __restrict__ gh,
                           const float* __restrict__ hprev,
                           float* __restrict__ h_out, _Float16* __restrict__ h16) {
  const int idx = blockIdx.x * blockDim.x + threadIdx.x;   // 0..65535
  const int n = idx >> 9, j = idx & 511;
  const float* gin = gi + n * 3 * DH;
  const float* ghn = gh + n * 3 * DH;
  const float rg = 1.f / (1.f + __expf(-(gin[j]        + ghn[j])));
  const float z  = 1.f / (1.f + __expf(-(gin[DH + j]   + ghn[DH + j])));
  const float ng = tanhf(gin[2 * DH + j] + rg * ghn[2 * DH + j]);
  const float h  = (1.f - z) * ng + z * hprev[idx];
  h_out[idx] = h;
  h16[idx]   = (_Float16)h;
}

__global__ void sigmoid_kernel(float* __restrict__ x, int n) {
  const int i = blockIdx.x * blockDim.x + threadIdx.x;
  if (i < n) x[i] = 1.f / (1.f + __expf(-x[i]));
}

// w[n,m] = exp(s - rowmax[n]) / rowsum[n]
__global__ void softmax_w_kernel(const float* __restrict__ s, const float* __restrict__ rowmax,
                                 const float* __restrict__ rowsum, float* __restrict__ w) {
  const int i = blockIdx.x * blockDim.x + threadIdx.x;     // 0..524287
  const int n = i >> 12;                                   // M=4096
  w[i] = __expf(s[i] - rowmax[n]) / rowsum[n];
}

// ---------------------------------------------------------------------------
// C_new = C*(1 - w*e) + w*v   streamed v4f; NT load (last use of C) + NT store
// keep the 512MB stream from thrashing L2.
// ---------------------------------------------------------------------------
__global__ void cnew_kernel(const float* __restrict__ C, const float* __restrict__ w,
                            const float* __restrict__ e, const float* __restrict__ v,
                            float* __restrict__ Cout) {
  const size_t t  = (size_t)blockIdx.x * blockDim.x + threadIdx.x;  // 16,777,216 threads
  const size_t nm = t >> 5;                 // D/4 = 32 v4f chunks per row
  const int    d4 = (int)(t & 31);
  const int    n  = (int)(nm >> 12);        // M = 4096
  const float  wv = w[nm];
  const v4f cv = __builtin_nontemporal_load(((const v4f*)C) + t);
  const v4f ev = ((const v4f*)e)[(n << 5) + d4];
  const v4f vv = ((const v4f*)v)[(n << 5) + d4];
  v4f o;
  o.x = cv.x * (1.f - wv * ev.x) + wv * vv.x;
  o.y = cv.y * (1.f - wv * ev.y) + wv * vv.y;
  o.z = cv.z * (1.f - wv * ev.z) + wv * vv.z;
  o.w = cv.w * (1.f - wv * ev.w) + wv * vv.w;
  __builtin_nontemporal_store(o, ((v4f*)Cout) + t);
}

// ---------------------------------------------------------------------------
extern "C" void kernel_launch(void* const* d_in, const int* in_sizes, int n_in,
                              void* d_out, int out_size, void* d_ws, size_t ws_size,
                              hipStream_t stream) {
  const float* h_prev = (const float*)d_in[0];
  const float* C      = (const float*)d_in[1];
  const float* Wk     = (const float*)d_in[2];
  const float* bk     = (const float*)d_in[3];
  const float* Wb     = (const float*)d_in[4];
  const float* bb     = (const float*)d_in[5];
  const float* We     = (const float*)d_in[6];
  const float* be     = (const float*)d_in[7];
  const float* Wv     = (const float*)d_in[8];
  const float* bv     = (const float*)d_in[9];
  const float* W_ih   = (const float*)d_in[10];
  const float* b_ih   = (const float*)d_in[11];
  const float* W_hh   = (const float*)d_in[12];
  const float* b_hh   = (const float*)d_in[13];

  // outputs flat in return order: h_o, C_new, k, r
  float* out   = (float*)d_out;
  float* out_h = out;                                  // 128*512
  float* out_C = out_h + N_B * DH;                     // 128*4096*128
  float* out_k = out_C + (size_t)N_B * M_SL * D_W;     // 128*128
  float* out_r = out_k + N_B * D_W;                    // 128*128

  // workspace layout: f32 region, then f16 region (16B-aligned boundary)
  float* ws      = (float*)d_ws;
  float* ws_gh   = ws;                       // 128*1536
  float* ws_gi   = ws_gh + N_B * 3 * DH;     // 128*1536
  float* ws_beta = ws_gi + N_B * 3 * DH;     // 128
  float* ws_rmax = ws_beta + N_B;            // 128
  float* ws_rsum = ws_rmax + N_B;            // 128
  float* ws_h    = ws_rsum + N_B;            // 128*512 (unused slot kept for align)
  float* ws_e    = ws_h + N_B * DH;          // 128*128
  float* ws_v    = ws_e + N_B * D_W;         // 128*128
  float* ws_s    = ws_v + N_B * D_W;         // 128*4096
  float* ws_w    = ws_s + (size_t)N_B * M_SL;// 128*4096
  _Float16* f16base = (_Float16*)(ws_w + (size_t)N_B * M_SL);
  _Float16* h16    = f16base;                  // 128*512   (h_prev)
  _Float16* ho16   = h16   + N_B * DH;         // 128*512   (h_o)
  _Float16* r16    = ho16  + N_B * DH;         // 128*128
  _Float16* Wk16   = r16   + N_B * D_W;        // 128*512
  _Float16* We16   = Wk16  + D_W * DH;         // 128*512
  _Float16* Wv16   = We16  + D_W * DH;         // 128*512
  _Float16* Wih16  = Wv16  + D_W * DH;         // 1536*128
  _Float16* Whh16  = Wih16 + 3 * DH * D_W;     // 1536*512

  // 0) one-time f32 -> f16 conversions (weights + h_prev)
  cvt_f16_kernel<<<(N_B * DH) / 256, 256, 0, stream>>>(h_prev, h16, N_B * DH);
  cvt_f16_kernel<<<(D_W * DH) / 256, 256, 0, stream>>>(Wk, Wk16, D_W * DH);
  cvt_f16_kernel<<<(D_W * DH) / 256, 256, 0, stream>>>(We, We16, D_W * DH);
  cvt_f16_kernel<<<(D_W * DH) / 256, 256, 0, stream>>>(Wv, Wv16, D_W * DH);
  cvt_f16_kernel<<<(3 * DH * D_W) / 256, 256, 0, stream>>>(W_ih, Wih16, 3 * DH * D_W);
  cvt_f16_kernel<<<(3 * DH * DH) / 256, 256, 0, stream>>>(W_hh, Whh16, 3 * DH * DH);

  // 1) k = h @ Wk.T + bk           (128x128x512, 64 tiles / 8 waves)
  gemm_xwt_wmma<<<8, 256, 0, stream>>>(h16, Wk16, bk, out_k, N_B, D_W, DH);
  // 2) gh = h @ W_hh.T + b_hh      (128x1536x512, 768 tiles)
  gemm_xwt_wmma<<<96, 256, 0, stream>>>(h16, Whh16, b_hh, ws_gh, N_B, 3 * DH, DH);
  // 3) beta
  beta_kernel<<<N_B, 128, 0, stream>>>(h_prev, Wb, bb, ws_beta);
  // 4) fused cosine-attention + online-softmax weighted read (C pass #1)
  flash_read_kernel<<<N_B, 256, 0, stream>>>(C, out_k, ws_beta, ws_s, ws_rmax, ws_rsum,
                                             out_r, r16);
  // 5) gi = r @ W_ih.T + b_ih      (128x1536x128)
  gemm_xwt_wmma<<<96, 256, 0, stream>>>(r16, Wih16, b_ih, ws_gi, N_B, 3 * DH, D_W);
  // 6) GRU update -> h_o (f32 out + f16 ws copy)
  gru_kernel<<<(N_B * DH) / 256, 256, 0, stream>>>(ws_gi, ws_gh, h_prev, out_h, ho16);
  // 7) e_raw, v
  gemm_xwt_wmma<<<8, 256, 0, stream>>>(ho16, We16, be, ws_e, N_B, D_W, DH);
  gemm_xwt_wmma<<<8, 256, 0, stream>>>(ho16, Wv16, bv, ws_v, N_B, D_W, DH);
  // 8) e = sigmoid(e_raw)
  sigmoid_kernel<<<(N_B * D_W + 255) / 256, 256, 0, stream>>>(ws_e, N_B * D_W);
  // 9) w from stored logits
  softmax_w_kernel<<<(N_B * M_SL) / 256, 256, 0, stream>>>(ws_s, ws_rmax, ws_rsum, ws_w);
  // 10) C_new erase/add (C pass #2: NT load + NT store)
  cnew_kernel<<<(unsigned)(((size_t)N_B * M_SL * D_W / 4) / 256), 256, 0, stream>>>(
      C, ws_w, ws_e, ws_v, out_C);
}
